// Attention_2422361554963
// MI455X (gfx1250) — matrix operand order repaired and checked by reading, not compiled, not run on hardware
//
#include <hip/hip_runtime.h>
#include <hip/hip_bf16.h>

// ---------------------------------------------------------------------------
// Banded additive attention for MI455X (gfx1250, wave32, WMMA).
//   q = x@Wt, k = x@Wx                      -> v_wmma_f32_16x16x32_f16
//   s = Wa . tanh(q_i + k_j + bh)           -> VALU (v_exp_f32 based tanh)
//   e = exp(sigmoid(s + ba)) * band * mask
//   a = e / (rowsum + eps)                  -> __shfl_xor wave32 reduction
//   v = a @ x                               -> v_wmma_f32_16x16x32_f16
// Weight staging via GLOBAL_LOAD_ASYNC_TO_LDS_B128 (ASYNCcnt) when available.
// ---------------------------------------------------------------------------

typedef _Float16 h16;
typedef __attribute__((ext_vector_type(16))) _Float16 v16h;
typedef __attribute__((ext_vector_type(8)))  _Float16 v8h;
typedef __attribute__((ext_vector_type(8)))  float    v8f;
typedef __attribute__((ext_vector_type(4)))  int      v4i;

#if defined(__has_builtin)
#if __has_builtin(__builtin_amdgcn_global_load_async_to_lds_b128) && \
    __has_builtin(__builtin_amdgcn_s_wait_asynccnt)
#define HAVE_ASYNC_LDS 1
#endif
#endif

#define L_DIM    2048
#define F_DIM    128
#define U_DIM    32
#define ROWS_PB  64          // queries per block (4 waves x 16 rows)
#define KEYS     192         // staged key rows per block: [q0-64, q0+128)
#define KEYS_PAD 208         // padded so B-fragments of the value GEMM stay in-bounds
#define KK_STR   36          // f32 row stride for kk/q (16B-aligned rows -> b128 reads)
#define A_STR    160         // a-tile row stride in halfs (5 chunks of 32 keys)

struct __align__(16) Smem {
    h16   WtT[U_DIM * F_DIM];     // Wt^T : [n][k], WMMA B-layout source
    h16   WxT[U_DIM * F_DIM];     // Wx^T : [n][k]
    h16   xT [F_DIM * KEYS_PAD];  // x^T  : [f][key] (f16), B operand of a@x
    union {                       // kk/qq are dead during phase 0 -> reuse as
        struct {                  // async-DMA staging area for raw f32 weights
            float kk[KEYS * KK_STR];     // k-projection of staged keys (f32)
            float qq[ROWS_PB * KK_STR];  // q-projection of block's queries (f32)
        } proj;
        float wstage[2 * U_DIM * F_DIM]; // [0..4095]=Wt, [4096..8191]=Wx (32 KB)
    };
    h16   aP [ROWS_PB * A_STR];   // normalized attention, WMMA A-layout source
    float kmask[KEYS];            // mask * in-range for staged keys
    float bh_s[U_DIM];
    float wa_s[U_DIM];
    float ba_s;
};

__device__ __forceinline__ float fast_tanh(float x) {
    float e2 = __expf(2.0f * x);                  // v_exp_f32
    return 1.0f - __fdividef(2.0f, e2 + 1.0f);    // exact at +/-inf saturation
}

// A fragment (16x32 f16) from a global f32 row (this lane's M row), ISA layout:
// halfs 0..7 -> K = k0 + lhi*8 + t ; halfs 8..15 -> K = k0 + 16 + lhi*8 + t
__device__ __forceinline__ v16h afrag_global(const float* __restrict__ rb,
                                             int k0, int lhi, bool valid) {
    union { v16h v; _Float16 h[16]; } u;
#pragma unroll
    for (int t = 0; t < 8; ++t) {
        u.h[t]     = valid ? (_Float16)rb[k0 + lhi * 8 + t]      : (_Float16)0.0f;
        u.h[t + 8] = valid ? (_Float16)rb[k0 + 16 + lhi * 8 + t] : (_Float16)0.0f;
    }
    return u.v;
}

// A fragment from LDS row-major [M][K]: two 16-byte runs, 16 halfs apart.
__device__ __forceinline__ v16h afrag_lds(const h16* base) {
    union { v16h v; v8h p[2]; } u;
    u.p[0] = *(const v8h*)(base);
    u.p[1] = *(const v8h*)(base + 16);
    return u.v;
}

// B fragment (32x16 f16) from LDS B^T row-major [N][K]: two 16-byte runs, 16 halfs apart.
__device__ __forceinline__ v16h bfrag_lds(const h16* base) {
    union { v16h v; v8h p[2]; } u;
    u.p[0] = *(const v8h*)(base);
    u.p[1] = *(const v8h*)(base + 16);
    return u.v;
}

__global__ __launch_bounds__(128)
void band_attn_kernel(const float* __restrict__ x,
                      const unsigned char* __restrict__ mask,
                      const float* __restrict__ Wt,
                      const float* __restrict__ Wx,
                      const float* __restrict__ bh,
                      const float* __restrict__ Wa,
                      const float* __restrict__ ba,
                      float* __restrict__ out) {
    __shared__ Smem sm;

    const int tid  = threadIdx.x;
    const int lane = tid & 31;
    const int wave = tid >> 5;
    const int lrow = lane & 15;   // N / M-row selector within fragment
    const int lhi  = lane >> 4;   // K-half selector
    const int b    = blockIdx.y;
    const int q0   = blockIdx.x * ROWS_PB;
    const int kstart = q0 - 64;   // first staged key (may be < 0)

    // ---------------- Phase 0: stage weights, x^T, masks ----------------
#if HAVE_ASYNC_LDS
    // Kick off async DMA of raw f32 Wt|Wx into the (currently dead) kk/qq
    // region: 2048 x 16B lane-transfers, tracked on ASYNCcnt, overlapped
    // with the x^T transpose/convert below.  (Toolchain declares the builtin
    // with generic v4i* pointers: (global src, lds dst, offset, cpol).)
    for (int c = tid; c < 2 * U_DIM * F_DIM / 4; c += 128) {
        const float* src = (c < 1024) ? (Wt + 4 * c) : (Wx + 4 * (c - 1024));
        __builtin_amdgcn_global_load_async_to_lds_b128(
            (v4i*)src, (v4i*)(sm.wstage + 4 * c), 0, 0);
    }
#endif
    // Prefetch this block's x rows (global_prefetch_b8); they are re-read by
    // the WMMA A-fragment builders in phase 1.
    for (int r = tid; r < KEYS; r += 128) {
        int g = kstart + r;
        if (g >= 0 && g < L_DIM)
            __builtin_prefetch(x + ((size_t)b * L_DIM + g) * F_DIM, 0, 3);
    }
    for (int r = tid; r < KEYS_PAD; r += 128) {
        int  g     = kstart + r;
        bool valid = (r < KEYS) && (g >= 0) && (g < L_DIM);
        const float* row = x + ((size_t)b * L_DIM + (valid ? g : 0)) * F_DIM;
#pragma unroll 4
        for (int f = 0; f < F_DIM; ++f)
            sm.xT[f * KEYS_PAD + r] = valid ? (h16)row[f] : (h16)0.0f;
    }
    for (int r = tid; r < KEYS; r += 128) {
        int g = kstart + r;
        sm.kmask[r] = (g >= 0 && g < L_DIM)
                        ? (mask[(size_t)b * L_DIM + g] ? 1.0f : 0.0f) : 0.0f;
    }
    if (tid < U_DIM) { sm.bh_s[tid] = bh[tid]; sm.wa_s[tid] = Wa[tid]; }
    if (tid == 0)    { sm.ba_s = ba[0]; }

#if HAVE_ASYNC_LDS
    __builtin_amdgcn_s_wait_asynccnt(0);   // own wave's DMA complete
    __syncthreads();                       // -> all waves' DMA complete
    for (int idx = tid; idx < U_DIM * F_DIM; idx += 128) {
        int k = idx >> 5, n = idx & 31;
        sm.WtT[n * F_DIM + k] = (h16)sm.wstage[idx];
        sm.WxT[n * F_DIM + k] = (h16)sm.wstage[U_DIM * F_DIM + idx];
    }
#else
    for (int idx = tid; idx < U_DIM * F_DIM; idx += 128) {
        int k = idx >> 5, n = idx & 31;
        sm.WtT[n * F_DIM + k] = (h16)Wt[idx];
        sm.WxT[n * F_DIM + k] = (h16)Wx[idx];
    }
#endif
    __syncthreads();

    // ---------------- Phase 1: kk = x_keys @ Wx, q = x_q @ Wt (WMMA) ----
    for (int tt = 0; tt < 3; ++tt) {                  // 12 key tiles / 4 waves
        const int  t     = wave * 3 + tt;
        const int  grow  = kstart + t * 16 + lrow;    // this lane's A row
        const bool valid = (grow >= 0) && (grow < L_DIM);
        const float* rb  = x + ((size_t)b * L_DIM + (valid ? grow : 0)) * F_DIM;
        v16h a4[4];
#pragma unroll
        for (int kc = 0; kc < 4; ++kc) a4[kc] = afrag_global(rb, kc * 32, lhi, valid);
#pragma unroll
        for (int nt = 0; nt < 2; ++nt) {
            v8f c = {};
#pragma unroll
            for (int kc = 0; kc < 4; ++kc) {
                v16h bf = bfrag_lds(&sm.WxT[(nt * 16 + lrow) * F_DIM + kc * 32 + lhi * 8]);
                c = __builtin_amdgcn_wmma_f32_16x16x32_f16(false, a4[kc], false, bf,
                                                           (short)0, c, false, false);
            }
#pragma unroll
            for (int r = 0; r < 8; ++r)
                sm.proj.kk[(t * 16 + lhi * 8 + r) * KK_STR + nt * 16 + lrow] = c[r];
        }
    }
    {   // query projection (queries are always in-range)
        const int grow  = q0 + wave * 16 + lrow;
        const float* rb = x + ((size_t)b * L_DIM + grow) * F_DIM;
        v16h a4[4];
#pragma unroll
        for (int kc = 0; kc < 4; ++kc) a4[kc] = afrag_global(rb, kc * 32, lhi, true);
#pragma unroll
        for (int nt = 0; nt < 2; ++nt) {
            v8f c = {};
#pragma unroll
            for (int kc = 0; kc < 4; ++kc) {
                v16h bf = bfrag_lds(&sm.WtT[(nt * 16 + lrow) * F_DIM + kc * 32 + lhi * 8]);
                c = __builtin_amdgcn_wmma_f32_16x16x32_f16(false, a4[kc], false, bf,
                                                           (short)0, c, false, false);
            }
#pragma unroll
            for (int r = 0; r < 8; ++r)
                sm.proj.qq[(wave * 16 + lhi * 8 + r) * KK_STR + nt * 16 + lrow] = c[r];
        }
    }
    __syncthreads();

    // ---------------- Phase 2: scores + normalization (VALU trans) ------
    const int kb = wave * 16;   // this wave's local key-window base
    for (int i = 0; i < 16; ++i) {
        const float4* qr4 = (const float4*)&sm.proj.qq[(wave * 16 + i) * KK_STR];
        float qb[U_DIM];
#pragma unroll
        for (int u4 = 0; u4 < U_DIM / 4; ++u4) {       // 8x ds_load_b128 (bcast)
            float4 t = qr4[u4];
            qb[4 * u4 + 0] = t.x + sm.bh_s[4 * u4 + 0];
            qb[4 * u4 + 1] = t.y + sm.bh_s[4 * u4 + 1];
            qb[4 * u4 + 2] = t.z + sm.bh_s[4 * u4 + 2];
            qb[4 * u4 + 3] = t.w + sm.bh_s[4 * u4 + 3];
        }
        const int   qlocal = 64 + wave * 16 + i;       // query row in key coords
        const float qm     = sm.kmask[qlocal];
        float ech[5];
        float psum = 0.0f;
#pragma unroll
        for (int kc = 0; kc < 5; ++kc) {
            int  jl   = kb + kc * 32 + lane;
            int  d    = jl - qlocal;
            bool band = (d >= -64) && (d <= 63);       // matrix_band_part(64, 63)
            int  jc   = band ? jl : 0;                 // safe clamp (e forced to 0)
            const float4* kr4 = (const float4*)&sm.proj.kk[jc * KK_STR];
            float kv[U_DIM];
#pragma unroll
            for (int u4 = 0; u4 < U_DIM / 4; ++u4) {   // 8x ds_load_b128
                float4 t = kr4[u4];
                kv[4 * u4 + 0] = t.x; kv[4 * u4 + 1] = t.y;
                kv[4 * u4 + 2] = t.z; kv[4 * u4 + 3] = t.w;
            }
            float s = 0.0f;
#pragma unroll
            for (int u = 0; u < U_DIM; ++u)
                s += fast_tanh(qb[u] + kv[u]) * sm.wa_s[u];
            float z   = s + sm.ba_s;
            float sig = __fdividef(1.0f, 1.0f + __expf(-z));
            float e   = __expf(sig) * (band ? 1.0f : 0.0f) * qm * sm.kmask[jc];
            ech[kc] = e;
            psum   += e;
        }
#pragma unroll
        for (int off = 16; off > 0; off >>= 1)         // wave32 row-sum
            psum += __shfl_xor(psum, off, 32);
        float inv = __fdividef(1.0f, psum + 1e-7f);
#pragma unroll
        for (int kc = 0; kc < 5; ++kc)
            sm.aP[(wave * 16 + i) * A_STR + kc * 32 + lane] = (h16)(ech[kc] * inv);
    }
    __syncthreads();

    // ---------------- Phase 3: v = a @ x (WMMA over key chunks) ---------
    v16h aF[5];
#pragma unroll
    for (int kc = 0; kc < 5; ++kc)
        aF[kc] = afrag_lds(&sm.aP[(wave * 16 + lrow) * A_STR + kc * 32 + lhi * 8]);
#pragma unroll
    for (int nt = 0; nt < 8; ++nt) {                   // 8 x 16 output columns
        v8f c = {};
#pragma unroll
        for (int kc = 0; kc < 5; ++kc) {
            v16h bf = bfrag_lds(&sm.xT[(nt * 16 + lrow) * KEYS_PAD + kb + kc * 32 + lhi * 8]);
            c = __builtin_amdgcn_wmma_f32_16x16x32_f16(false, aF[kc], false, bf,
                                                       (short)0, c, false, false);
        }
#pragma unroll
        for (int r = 0; r < 8; ++r) {
            int qi  = q0 + wave * 16 + lhi * 8 + r;
            int col = nt * 16 + lrow;
            out[((size_t)b * L_DIM + qi) * F_DIM + col] = c[r];
        }
    }
}

extern "C" void kernel_launch(void* const* d_in, const int* in_sizes, int n_in,
                              void* d_out, int out_size, void* d_ws, size_t ws_size,
                              hipStream_t stream) {
    const float*         x    = (const float*)d_in[0];
    const unsigned char* mask = (const unsigned char*)d_in[1];
    const float*         Wt   = (const float*)d_in[2];
    const float*         Wx   = (const float*)d_in[3];
    const float*         bh   = (const float*)d_in[4];
    const float*         Wa   = (const float*)d_in[5];
    const float*         ba   = (const float*)d_in[6];
    float*               out  = (float*)d_out;

    const int B = in_sizes[0] / (L_DIM * F_DIM);     // = 2
    dim3 grid(L_DIM / ROWS_PB, B);
    band_attn_kernel<<<grid, 128, 0, stream>>>(x, mask, Wt, Wx, bh, Wa, ba, out);
}
